// SparseEncoder_35003983462792
// MI455X (gfx1250) — matrix operand order repaired and checked by reading, compiled, not run on hardware
//
#include <hip/hip_runtime.h>
#include <hip/hip_bf16.h>
#include <stdint.h>

// ---------------------------------------------------------------------------
// SparseEncoder for MI455X (gfx1250): bf16 implicit-GEMM convs on the WMMA
// pipe (v_wmma_f32_16x16x32_bf16), TDM (tensor_load_to_lds) staged weight
// tiles tracked by TENSORcnt, 2 B-tiles per A fragment, zero-guard row for
// padding (branchless A loads), software-pipelined A with split tail, fused
// BN+mask+residual+ReLU epilogue, masked maxpools writing skips in f32.
// ---------------------------------------------------------------------------

typedef __attribute__((ext_vector_type(16))) __bf16 v16bf;
typedef __attribute__((ext_vector_type(8)))  float  v8f;
typedef __attribute__((ext_vector_type(4)))  unsigned int v4u;
typedef __attribute__((ext_vector_type(8)))  int    v8i;
typedef __attribute__((ext_vector_type(4)))  int    v4i;
typedef unsigned short u16;
typedef unsigned char  u8;

__device__ __forceinline__ u16 f2bf(float f) {
  union { float f; uint32_t u; } x; x.f = f;
  uint32_t lsb = (x.u >> 16) & 1u;
  x.u += 0x7fffu + lsb;                 // round-to-nearest-even
  return (u16)(x.u >> 16);
}
__device__ __forceinline__ float bf2f(u16 h) {
  union { uint32_t u; float f; } x; x.u = ((uint32_t)h) << 16;
  return x.f;
}

union Frag16 { v16bf v; uint4 q[2]; };

// ---------------------------------------------------------------------------
// pack: f32 NHWC -> bf16 NHWC, derive activity mask (any channel != 0)
// ---------------------------------------------------------------------------
__global__ void pack_x_kernel(const float* __restrict__ x, u16* __restrict__ xb,
                              u8* __restrict__ m, int total, int C) {
  int s = blockIdx.x * 256 + threadIdx.x;
  if (s >= total) return;
  const float* px = x + (size_t)s * C;
  u16* pb = xb + (size_t)s * C;
  bool any = false;
  for (int c = 0; c < C; c += 2) {
    float a = px[c], b = px[c + 1];
    any |= (a != 0.0f) | (b != 0.0f);
    uint32_t u = (uint32_t)f2bf(a) | ((uint32_t)f2bf(b) << 16);
    *(uint32_t*)(pb + c) = u;
  }
  m[s] = any ? 1 : 0;
}

__global__ void zero_u16_kernel(u16* __restrict__ p, int n) {
  int i = blockIdx.x * 256 + threadIdx.x;
  if (i < n) p[i] = 0;
}

// ---------------------------------------------------------------------------
// weight repack: HWIO f32 [k,k,Cin,Cout] -> bf16 [tap][Cout][Cin]
// ---------------------------------------------------------------------------
__global__ void repack_w_kernel(const float* __restrict__ w, u16* __restrict__ wp,
                                int taps, int Cin, int Cout) {
  int i = blockIdx.x * 256 + threadIdx.x;
  int tot = taps * Cin * Cout;
  if (i >= tot) return;
  int ci = i % Cin; int r = i / Cin; int co = r % Cout; int tp = r / Cout;
  wp[i] = f2bf(w[((size_t)tp * Cin + ci) * Cout + co]);
}

// ---------------------------------------------------------------------------
// BN fuse: scale = gamma*rsqrt(var+eps), bias = beta - mu*scale
// ---------------------------------------------------------------------------
__global__ void bnfuse_kernel(const float* __restrict__ g, const float* __restrict__ be,
                              const float* __restrict__ mu, const float* __restrict__ var,
                              float* __restrict__ sc, float* __restrict__ bi, int C) {
  int c = blockIdx.x * 64 + threadIdx.x;
  if (c >= C) return;
  float s = g[c] * rsqrtf(var[c] + 1e-5f);
  sc[c] = s;
  bi[c] = be[c] - mu[c] * s;
}

// ---------------------------------------------------------------------------
// Implicit-GEMM conv + fused BN*mask [+res] + ReLU.  B = 4 fixed.
// Workgroup: 256 threads = 8 waves; tile = 64 positions x 64 out-channels.
//   posTile = wave>>1 (4 tiles of 16 consecutive x positions)
//   chGrp   = wave&1  (which 32-ch half; each wave owns TWO 16-ch B tiles)
// Weights [tap][Cout][Cin] bf16; the 64xCin slab for this block's 64 out-ch
// is DMA'd into LDS per tap by the Tensor Data Mover (one descriptor issued
// by wave 0, completion via s_wait_tensorcnt, then workgroup barrier).  Each
// wave reuses one A fragment for two WMMAs (acc0/acc1).  Out-of-bounds taps
// read a zero guard row; the last K chunk is split out of the loop so the
// steady state is branchless with the next A fragment always in flight.
// ---------------------------------------------------------------------------
__global__ __launch_bounds__(256) void conv_bn_relu_wmma(
    const u16* __restrict__ in, const u16* __restrict__ wgt,
    const float* __restrict__ scale, const float* __restrict__ bias,
    const u8* __restrict__ mask, const u16* __restrict__ res,
    const u16* __restrict__ zrow,
    u16* __restrict__ out, int H, int W, int Cin, int Cout, int ksz) {
  __shared__ u16 lds_w[64 * 256];          // 64 out-ch x (Cin<=256) bf16 = 32 KB

  const int tid     = threadIdx.x;
  const int lane    = tid & 31;
  const int wave    = tid >> 5;
  const int chGrp   = wave & 1;
  const int posTile = wave >> 1;
  const int n     = lane & 15;             // A row / B column within tile
  const int khalf = lane >> 4;

  const int tilesPerRow = W >> 6;          // 64 positions per workgroup row-tile
  const int tx = blockIdx.x % tilesPerRow;
  const int yb = blockIdx.x / tilesPerRow;
  const int y  = yb % H;
  const int b  = yb / H;
  const int x0 = (tx << 6) + (posTile << 4);
  const int co0 = blockIdx.y << 6;
  const int pad = (ksz - 1) >> 1;
  const int taps = ksz * ksz;

  v8f acc0 = {}, acc1 = {};

  for (int t = 0; t < taps; ++t) {
    __syncthreads();                       // all waves done reading previous slab
    const u16* wsrc = wgt + ((size_t)t * Cout + co0) * (size_t)Cin;
    if (tid < 32) {                        // wave 0 issues the TDM descriptor
      // --- Tensor DMA descriptor (D#), 2-D tile: 64 rows x Cin bf16 -------
      const uint32_t ldsAddr = (uint32_t)(uintptr_t)(void*)lds_w;  // LDS offset
      const uint64_t ga = (uint64_t)(uintptr_t)(const void*)wsrc;  // tile start
      v4u g0;
      g0[0] = 1u;                                          // count=1 (valid D#)
      g0[1] = ldsAddr;                                     // lds_addr (bytes)
      g0[2] = (uint32_t)ga;                                // global_addr[31:0]
      g0[3] = (uint32_t)((ga >> 32) & 0x01FFFFFFu)         // global_addr[56:32]
              | (2u << 30);                                // type=2 ("image")
      v8i g1;
      g1[0] = (int)(1u << 16);                             // data_size=1 (2 B)
      g1[1] = (int)(((uint32_t)Cin & 0xFFFFu) << 16);      // tensor_dim0 lo16
      g1[2] = (int)((((uint32_t)Cin >> 16) & 0xFFFFu)      // tensor_dim0 hi16
              | (64u << 16));                              // tensor_dim1 lo16 (=64 rows)
      g1[3] = (int)(((uint32_t)Cin & 0xFFFFu) << 16);      // dim1 hi=0 | tile_dim0=Cin
      g1[4] = 64;                                          // tile_dim1=64, tile_dim2=0
      g1[5] = Cin;                                         // tensor_dim0_stride lo32
      g1[6] = 0;                                           // stride hi | dim1_stride lo
      g1[7] = 0;                                           // dim1_stride hi
      v4i g2 = {0, 0, 0, 0};                               // 2-D: groups 2/3 unused
      v4i g3 = {0, 0, 0, 0};
      v8i g4 = {0, 0, 0, 0, 0, 0, 0, 0};                   // extra group (unused)
      __builtin_amdgcn_tensor_load_to_lds(g0, g1, g2, g3, g4, 0);
      __builtin_amdgcn_s_wait_tensorcnt(0);
    }
    if (t + 1 < taps) {                    // prefetch next tap's weight slab
      const char* nxt = (const char*)(wgt + ((size_t)(t + 1) * Cout + co0) * (size_t)Cin);
      if (tid * 128 < 64 * Cin * 2) __builtin_prefetch(nxt + tid * 128, 0, 0);
    }
    __syncthreads();                       // slab visible to all waves

    const int dy = t / ksz - pad;
    const int dx = t % ksz - pad;
    const int yi = y + dy;
    const int xi = x0 + n + dx;            // this lane's A row position
    const bool valid = (yi >= 0) && (yi < H) && (xi >= 0) && (xi < W);
    // Branchless padding: invalid rows read a zeroed guard row.
    const u16* arow = valid ? (in + ((size_t)(b * H + yi) * W + xi) * (size_t)Cin)
                            : zrow;
    const u16* brow = lds_w + (size_t)((chGrp << 5) + n) * (size_t)Cin + (khalf << 4);
    const int bstride16 = 16 * Cin;        // second 16-ch B tile

    Frag16 a;                              // pipelined A fragment
    a.q[0] = *(const uint4*)(arow + (khalf << 4));        // K 0..7 / 16..23
    a.q[1] = *(const uint4*)(arow + 8 + (khalf << 4));    // K 8..15 / 24..31

    int cc = 0;
    for (; cc + 32 < Cin; cc += 32) {      // steady state: branchless, A in flight
      Frag16 an;
      an.q[0] = *(const uint4*)(arow + cc + 32 + (khalf << 4));
      an.q[1] = *(const uint4*)(arow + cc + 40 + (khalf << 4));
      Frag16 b0, b1;
      b0.q[0] = *(const uint4*)(brow + cc);                // ds_load_b128
      b0.q[1] = *(const uint4*)(brow + cc + 8);
      b1.q[0] = *(const uint4*)(brow + bstride16 + cc);
      b1.q[1] = *(const uint4*)(brow + bstride16 + cc + 8);
      acc0 = __builtin_amdgcn_wmma_f32_16x16x32_bf16(
          false, a.v, false, b0.v, (short)0, acc0, false, false);
      acc1 = __builtin_amdgcn_wmma_f32_16x16x32_bf16(
          false, a.v, false, b1.v, (short)0, acc1, false, false);
      a = an;
    }
    {                                       // final K chunk (no A preload)
      Frag16 b0, b1;
      b0.q[0] = *(const uint4*)(brow + cc);
      b0.q[1] = *(const uint4*)(brow + cc + 8);
      b1.q[0] = *(const uint4*)(brow + bstride16 + cc);
      b1.q[1] = *(const uint4*)(brow + bstride16 + cc + 8);
      acc0 = __builtin_amdgcn_wmma_f32_16x16x32_bf16(
          false, a.v, false, b0.v, (short)0, acc0, false, false);
      acc1 = __builtin_amdgcn_wmma_f32_16x16x32_bf16(
          false, a.v, false, b1.v, (short)0, acc1, false, false);
    }
  }

  // Epilogue: D layout — VGPR r: lanes 0-15 -> M=r, lanes 16-31 -> M=8+r; N=lane&15.
  const int coA = co0 + (chGrp << 5) + n;
  const int coB = coA + 16;
  const float scA = scale[coA], biA = bias[coA];
  const float scB = scale[coB], biB = bias[coB];
#pragma unroll
  for (int r = 0; r < 8; ++r) {
    const int mrow = r + (khalf << 3);
    const size_t p = (size_t)(b * H + y) * W + (x0 + mrow);
    const bool act = mask[p] != 0;
    float v0 = acc0[r] * scA + biA;
    float v1 = acc1[r] * scB + biB;
    v0 = act ? v0 : 0.0f;                            // BN re-mask
    v1 = act ? v1 : 0.0f;
    if (res) {                                       // residual (0 at inactive)
      v0 += bf2f(res[p * (size_t)Cout + coA]);
      v1 += bf2f(res[p * (size_t)Cout + coB]);
    }
    out[p * (size_t)Cout + coA] = f2bf(fmaxf(v0, 0.0f));   // ReLU
    out[p * (size_t)Cout + coB] = f2bf(fmaxf(v1, 0.0f));
  }
}

// ---------------------------------------------------------------------------
// Masked 3x3/stride2/pad1 maxpool.  Post-ReLU values >= 0 and inactive sites
// hold 0, so max over the window equals max over active sites; new mask = any
// active in window.  Optionally writes bf16 (next stage) and f32 (skip out).
// ---------------------------------------------------------------------------
__global__ void pool_kernel(const u16* __restrict__ in, const u8* __restrict__ min_,
                            u16* __restrict__ outb, float* __restrict__ outf,
                            u8* __restrict__ mout, int H, int W, int C) {
  const int Ho = H >> 1, Wo = W >> 1;
  const size_t total = (size_t)4 * Ho * Wo * C;
  size_t t = (size_t)blockIdx.x * 256 + threadIdx.x;
  if (t >= total) return;
  int c = (int)(t % C);
  size_t s = t / C;
  int xo = (int)(s % Wo);
  int tmp = (int)(s / Wo);
  int yo = tmp % Ho;
  int b  = tmp / Ho;
  float best = 0.0f;
  bool any = false;
  for (int dy = -1; dy <= 1; ++dy) {
    int yi = 2 * yo + dy;
    if (yi < 0 || yi >= H) continue;
    for (int dx = -1; dx <= 1; ++dx) {
      int xi = 2 * xo + dx;
      if (xi < 0 || xi >= W) continue;
      size_t si = (size_t)(b * H + yi) * W + xi;
      if (min_[si]) {
        any = true;
        best = fmaxf(best, bf2f(in[si * C + c]));
      }
    }
  }
  float o = any ? best : 0.0f;
  if (outb) outb[s * C + c] = f2bf(o);
  if (outf) outf[s * C + c] = o;
  if (c == 0 && mout) mout[s] = any ? 1 : 0;
}

// ---------------------------------------------------------------------------
// Host orchestration
// ---------------------------------------------------------------------------
// d_in layout assumption (jax pytree flattening: dicts by sorted key, bn
// tuples in (gamma,beta,mu,var) order):
//   0:x  1:mask
//   2..5:   bn0 (g,b,m,v)
//   6..15:  l1b1  (bn1 g,b,m,v | bn2 g,b,m,v | w1 | w2)
//   16..25: l1b2   26..35: l2b1   36..45: l2b2   46..55: l3b1   56..65: l3b2
//   66: w0
extern "C" void kernel_launch(void* const* d_in, const int* in_sizes, int n_in,
                              void* d_out, int out_size, void* d_ws, size_t ws_size,
                              hipStream_t stream) {
  (void)in_sizes; (void)n_in; (void)out_size; (void)ws_size;
  const float* X = (const float*)d_in[0];
  float* OUT = (float*)d_out;

  // d_out offsets (floats): final | x | skip2 | skip3
  const size_t OFF_FINAL = 0;
  const size_t OFF_X     = (size_t)4 * 32 * 32 * 256;              // 1,048,576
  const size_t OFF_SKIP2 = OFF_X + (size_t)4 * 256 * 256 * 128;    // 34,603,008
  const size_t OFF_SKIP3 = OFF_SKIP2 + (size_t)4 * 128 * 128 * 64; // 38,797,312

  // workspace bump allocator (256B aligned)
  char* ws = (char*)d_ws;
  size_t off = 0;
  auto alloc = [&](size_t bytes) -> void* {
    void* p = ws + off;
    off = (off + bytes + 255) & ~(size_t)255;
    return p;
  };
  u16* S0 = (u16*)alloc((size_t)4 * 256 * 256 * 128 * 2); // 64 MB (xbf; later pooled stages)
  u16* S1 = (u16*)alloc((size_t)4 * 256 * 256 * 64 * 2);  // 32 MB rotating slots
  u16* S2 = (u16*)alloc((size_t)4 * 256 * 256 * 64 * 2);
  u16* S3 = (u16*)alloc((size_t)4 * 256 * 256 * 64 * 2);
  u16* WB = (u16*)alloc((size_t)9 * 256 * 256 * 2);       // packed weights (max l3 conv)
  float* SC = (float*)alloc(256 * 4);
  float* BI = (float*)alloc(256 * 4);
  u8* M0 = (u8*)alloc((size_t)4 * 256 * 256);
  u8* M1 = (u8*)alloc((size_t)4 * 128 * 128);
  u8* M2 = (u8*)alloc((size_t)4 * 64 * 64);
  u16* ZR = (u16*)alloc(256 * 2);                         // zero guard row (Cin<=256)

  auto run_conv = [&](const u16* in, const u8* msk, const u16* res, u16* out,
                      int H, int W, int Cin, int Cout, int k,
                      int w_idx, int bn_idx) {
    const float* w = (const float*)d_in[w_idx];
    int welems = k * k * Cin * Cout;
    repack_w_kernel<<<(welems + 255) / 256, 256, 0, stream>>>(w, WB, k * k, Cin, Cout);
    bnfuse_kernel<<<(Cout + 63) / 64, 64, 0, stream>>>(
        (const float*)d_in[bn_idx], (const float*)d_in[bn_idx + 1],
        (const float*)d_in[bn_idx + 2], (const float*)d_in[bn_idx + 3], SC, BI, Cout);
    dim3 grid(4 * H * (W / 64), Cout / 64);
    conv_bn_relu_wmma<<<grid, 256, 0, stream>>>(in, WB, SC, BI, msk, res, ZR, out,
                                                H, W, Cin, Cout, k);
  };
  auto run_pool = [&](const u16* in, const u8* min_, u16* outb, float* outf,
                      u8* mout, int H, int W, int C) {
    size_t total = (size_t)4 * (H / 2) * (W / 2) * C;
    pool_kernel<<<(unsigned)((total + 255) / 256), 256, 0, stream>>>(
        in, min_, outb, outf, mout, H, W, C);
  };

  // 0) zero guard row
  zero_u16_kernel<<<1, 256, 0, stream>>>(ZR, 256);
  // 1) pack input to bf16 + derive mask
  pack_x_kernel<<<(4 * 256 * 256 + 255) / 256, 256, 0, stream>>>(X, S0, M0, 4 * 256 * 256, 128);
  // raw x passthrough output
  (void)hipMemcpyAsync(OUT + OFF_X, X, (size_t)4 * 256 * 256 * 128 * sizeof(float),
                       hipMemcpyDeviceToDevice, stream);

  // 2) first_conv 7x7 128->64 + bn0 + relu
  run_conv(S0, M0, nullptr, S1, 256, 256, 128, 64, 7, /*w0*/66, /*bn0*/2);
  // layer_1 (residual blocks at 64ch)
  run_conv(S1, M0, nullptr, S2, 256, 256, 64, 64, 3, 14, 6);   // l1b1 conv1
  run_conv(S2, M0, S1,      S3, 256, 256, 64, 64, 3, 15, 10);  // l1b1 conv2 (+res)
  run_conv(S3, M0, nullptr, S1, 256, 256, 64, 64, 3, 24, 16);  // l1b2 conv1
  run_conv(S1, M0, S3,      S2, 256, 256, 64, 64, 3, 25, 20);  // l1b2 conv2 (+res)
  // pool1 -> 128x128, writes skip2 (f32)
  run_pool(S2, M0, S0, OUT + OFF_SKIP2, M1, 256, 256, 64);

  // layer_2 (64->128 expand, then residual)
  run_conv(S0, M1, nullptr, S1, 128, 128, 64, 128, 3, 34, 26); // l2b1 conv1
  run_conv(S1, M1, nullptr, S2, 128, 128, 128, 128, 3, 35, 30);// l2b1 conv2 (no res)
  run_conv(S2, M1, nullptr, S3, 128, 128, 128, 128, 3, 44, 36);// l2b2 conv1
  run_conv(S3, M1, S2,      S1, 128, 128, 128, 128, 3, 45, 40);// l2b2 conv2 (+res)
  // pool2 -> 64x64, writes skip3 (f32)
  run_pool(S1, M1, S0, OUT + OFF_SKIP3, M2, 128, 128, 128);

  // layer_3 (128->256 expand, then residual)
  run_conv(S0, M2, nullptr, S1, 64, 64, 128, 256, 3, 54, 46);  // l3b1 conv1
  run_conv(S1, M2, nullptr, S2, 64, 64, 256, 256, 3, 55, 50);  // l3b1 conv2 (no res)
  run_conv(S2, M2, nullptr, S3, 64, 64, 256, 256, 3, 64, 56);  // l3b2 conv1
  run_conv(S3, M2, S2,      S1, 64, 64, 256, 256, 3, 65, 60);  // l3b2 conv2 (+res)
  // pool3 -> 32x32, writes final (f32 only)
  run_pool(S1, M2, nullptr, OUT + OFF_FINAL, nullptr, 64, 64, 256);
}
